// BinnedRegressionInteractionGNN_67010079752304
// MI455X (gfx1250) — matrix operand order repaired and checked by reading, not compiled, not run on hardware
//
#include <hip/hip_runtime.h>

// ---------------------------------------------------------------------------
// Types for CDNA5 WMMA (wave32)
// ---------------------------------------------------------------------------
typedef __attribute__((ext_vector_type(16))) __bf16         v16bf;
typedef __attribute__((ext_vector_type(8)))  unsigned int   v8u;   // 32B raw
typedef __attribute__((ext_vector_type(8)))  float          v8f;

#define LDST 136   // padded row stride (ushorts) to avoid LDS bank conflicts

__device__ __forceinline__ unsigned short f32_to_bf16(float f) {
  unsigned int u = __builtin_bit_cast(unsigned int, f);
  u += 0x7fffu + ((u >> 16) & 1u);          // round-to-nearest-even
  return (unsigned short)(u >> 16);
}
__device__ __forceinline__ float bf16_to_f32(unsigned short h) {
  unsigned int u = ((unsigned int)h) << 16;
  return __builtin_bit_cast(float, u);
}
__device__ __forceinline__ unsigned int pack_bf16x2(float a, float b) {
  return (unsigned int)f32_to_bf16(a) | ((unsigned int)f32_to_bf16(b) << 16);
}
__device__ __forceinline__ void atomic_add_f32(float* p, float v) {
  atomicAdd(p, v);
}

// ---------------------------------------------------------------------------
// Weight shuffle: f32 row-major [K,128] -> bf16 WMMA B-fragment order.
// Fragment (c = 32-K chunk, t = 16-col tile): [c][t][lane][16 bf16], where
// lane<16 holds col=lane, K = chunk+0..7,16..23; lane>=16 holds col=lane-16,
// K = chunk+8..15,24..31 (2 values per VGPR).
// ---------------------------------------------------------------------------
__global__ void shuffle_w(const float* __restrict__ W, unsigned short* __restrict__ out,
                          int total) {
  int id = blockIdx.x * 256 + threadIdx.x;
  if (id >= total) return;
  int j    = id & 15;
  int lane = (id >> 4) & 31;
  int ft   = id >> 9;
  int t    = ft & 7;
  int c    = ft >> 3;
  int k    = c * 32 + (lane >> 4) * 8 + (j >> 3) * 16 + (j & 7);
  int col  = t * 16 + (lane & 15);
  out[id] = f32_to_bf16(W[k * 128 + col]);
}

// ---------------------------------------------------------------------------
// Node encoder (K=3 -> 128, LN, ReLU, 128 -> 128). VALU; one wave per node.
// ---------------------------------------------------------------------------
__global__ void __launch_bounds__(256)
node_encoder(const float* __restrict__ x,
             const float* __restrict__ W1, const float* __restrict__ b1,
             const float* __restrict__ g,  const float* __restrict__ be,
             const float* __restrict__ W2, const float* __restrict__ b2,
             unsigned short* __restrict__ x0, int N) {
  __shared__ float a_sh[8][128];
  const int wave = threadIdx.x >> 5, lane = threadIdx.x & 31;
  int node = blockIdx.x * 8 + wave;
  if (node >= N) node = N - 1;                       // benign duplicate work
  const float xv0 = x[node * 3 + 0], xv1 = x[node * 3 + 1], xv2 = x[node * 3 + 2];
  float h[4]; float s = 0.f, sq = 0.f;
  #pragma unroll
  for (int q = 0; q < 4; ++q) {
    int j = lane * 4 + q;
    float v = xv0 * W1[j] + xv1 * W1[128 + j] + xv2 * W1[256 + j] + b1[j];
    h[q] = v; s += v; sq += v * v;
  }
  #pragma unroll
  for (int off = 16; off; off >>= 1) { s += __shfl_xor(s, off); sq += __shfl_xor(sq, off); }
  const float m = s * (1.f / 128.f);
  const float r = rsqrtf(sq * (1.f / 128.f) - m * m + 1e-5f);
  #pragma unroll
  for (int q = 0; q < 4; ++q) {
    int j = lane * 4 + q;
    float v = (h[q] - m) * r * g[j] + be[j];
    a_sh[wave][j] = v > 0.f ? v : 0.f;
  }
  __syncthreads();
  float o[4];
  #pragma unroll
  for (int q = 0; q < 4; ++q) o[q] = b2[lane * 4 + q];
  for (int k = 0; k < 128; ++k) {
    float av = a_sh[wave][k];
    #pragma unroll
    for (int q = 0; q < 4; ++q) o[q] += av * W2[k * 128 + lane * 4 + q];
  }
  #pragma unroll
  for (int q = 0; q < 4; ++q) x0[node * 128 + lane * 4 + q] = f32_to_bf16(o[q]);
}

// ---------------------------------------------------------------------------
// A-fragment segment loader (branch-free after constant folding):
// 8 contiguous bf16 input features starting at (multiple-of-8) column k.
// ---------------------------------------------------------------------------
template <int MODE>
__device__ __forceinline__ uint4 load_seg(int r, int sN, int dN,
                                          const unsigned short* __restrict__ xsrc,
                                          const unsigned short* __restrict__ efeat,
                                          const float* __restrict__ msg, int k) {
  if (MODE == 1) {
    if (k < 128) return *(const uint4*)(xsrc + (size_t)r * 128 + k);
    const float4 a = *(const float4*)(msg + (size_t)r * 128 + (k - 128));
    const float4 b = *(const float4*)(msg + (size_t)r * 128 + (k - 128) + 4);
    uint4 o;
    o.x = pack_bf16x2(a.x, a.y); o.y = pack_bf16x2(a.z, a.w);
    o.z = pack_bf16x2(b.x, b.y); o.w = pack_bf16x2(b.z, b.w);
    return o;
  } else {
    if (MODE == 2 && k >= 256)
      return *(const uint4*)(efeat + (size_t)r * 128 + (k - 256));
    int node = (k < 128) ? sN : dN;
    return *(const uint4*)(xsrc + (size_t)node * 128 + (k & 127));
  }
}

__device__ __forceinline__ v8u seg_pair(uint4 s0, uint4 s1) {
  v8u a;
  a[0] = s0.x; a[1] = s0.y; a[2] = s0.z; a[3] = s0.w;
  a[4] = s1.x; a[5] = s1.y; a[6] = s1.z; a[7] = s1.w;
  return a;
}

// In-register LayerNorm + ReLU over one 16x128 C-layout tile, then bf16
// transpose-staging into LDS (row-major) for the second GEMM's A fragments.
__device__ __forceinline__ void ln_relu_to_lds(v8f (&acc)[8],
                                               unsigned short (*ldsw)[LDST],
                                               int tilebase,
                                               const float* bv, const float* gv,
                                               const float* bev, int l16, int hf) {
  #pragma unroll
  for (int rr = 0; rr < 8; ++rr) {
    float s = 0.f, sq = 0.f;
    #pragma unroll
    for (int t = 0; t < 8; ++t) {
      float v = acc[t][rr] + bv[t];
      acc[t][rr] = v; s += v; sq += v * v;
    }
    s += __shfl_xor(s, 1);  sq += __shfl_xor(sq, 1);
    s += __shfl_xor(s, 2);  sq += __shfl_xor(sq, 2);
    s += __shfl_xor(s, 4);  sq += __shfl_xor(sq, 4);
    s += __shfl_xor(s, 8);  sq += __shfl_xor(sq, 8);
    const float mn = s * (1.f / 128.f);
    const float rs = rsqrtf(sq * (1.f / 128.f) - mn * mn + 1e-5f);
    const int row = tilebase + hf * 8 + rr;
    #pragma unroll
    for (int t = 0; t < 8; ++t) {
      float v = (acc[t][rr] - mn) * rs * gv[t] + bev[t];
      v = v > 0.f ? v : 0.f;
      ldsw[row][t * 16 + l16] = f32_to_bf16(v);
    }
  }
}

// ---------------------------------------------------------------------------
// Fused MLP2 with WMMA:  out = relu(LN(in @ W1 + b1)) @ W2 + b2
// MODE 0: in = [x[start], x[end]]          (KIN=256)  rows = edges
// MODE 1: in = [x[row],   msg[row](f32)]   (KIN=256)  rows = nodes
// MODE 2: in = [x[start], x[end], e[row]]  (KIN=384)  rows = edges (in-place e)
// One wave computes a 32-row x 128-col output (two M-tiles sharing every
// B fragment). Block = 8 waves = 256 rows. Fully unrolled chunk loops.
// ---------------------------------------------------------------------------
template <int MODE, int KIN>
__global__ void __launch_bounds__(256, 1)
mlp2_wmma(int nrows,
          const unsigned short* __restrict__ xsrc,
          const unsigned short* __restrict__ efeat,
          const float* __restrict__ msg,
          const int* __restrict__ startI, const int* __restrict__ endI,
          const unsigned short* __restrict__ w1s, const float* __restrict__ b1,
          const float* __restrict__ g,  const float* __restrict__ be,
          const unsigned short* __restrict__ w2s, const float* __restrict__ b2,
          unsigned short* __restrict__ out) {
  constexpr int NCH = KIN / 32;
  __shared__ __align__(16) unsigned short lds_a[8][32][LDST];

  const int wave = threadIdx.x >> 5;
  const int lane = threadIdx.x & 31;
  const int hf   = lane >> 4;
  const int l16  = lane & 15;
  const int rowbase = (blockIdx.x * 8 + wave) * 32;

  int r0 = rowbase + l16;       if (r0 >= nrows) r0 = nrows - 1;
  int r1 = rowbase + 16 + l16;  if (r1 >= nrows) r1 = nrows - 1;
  int sN0 = 0, dN0 = 0, sN1 = 0, dN1 = 0;
  if (MODE != 1) {
    sN0 = startI[r0]; dN0 = endI[r0];
    sN1 = startI[r1]; dN1 = endI[r1];
  }

  const int kb = hf * 8;             // A-fragment K offset per half-wave

  v8f acc0[8], acc1[8];
  #pragma unroll
  for (int t = 0; t < 8; ++t) { acc0[t] = (v8f)0.0f; acc1[t] = (v8f)0.0f; }

  // ---------------- GEMM1: in[32 x KIN] @ W1[KIN x 128] ----------------
  #pragma unroll
  for (int c = 0; c < NCH; ++c) {
    const int k0 = c * 32 + kb;
    v16bf A0 = __builtin_bit_cast(v16bf,
        seg_pair(load_seg<MODE>(r0, sN0, dN0, xsrc, efeat, msg, k0),
                 load_seg<MODE>(r0, sN0, dN0, xsrc, efeat, msg, k0 + 16)));
    v16bf A1 = __builtin_bit_cast(v16bf,
        seg_pair(load_seg<MODE>(r1, sN1, dN1, xsrc, efeat, msg, k0),
                 load_seg<MODE>(r1, sN1, dN1, xsrc, efeat, msg, k0 + 16)));
    const unsigned short* wp = w1s + (size_t)c * (8 * 32 * 16) + (size_t)lane * 16;
    if (c + 1 < NCH)
      __builtin_prefetch(wp + (size_t)(8 * 32 * 16), 0, 1);
    #pragma unroll
    for (int h2 = 0; h2 < 2; ++h2) {
      v8u bu[4];
      #pragma unroll
      for (int t = 0; t < 4; ++t)
        bu[t] = *(const v8u*)(wp + (size_t)(h2 * 4 + t) * (32 * 16));
      #pragma unroll
      for (int t = 0; t < 4; ++t) {
        v16bf B = __builtin_bit_cast(v16bf, bu[t]);
        acc0[h2 * 4 + t] = __builtin_amdgcn_wmma_f32_16x16x32_bf16(
            false, A0, false, B, (short)0, acc0[h2 * 4 + t], false, false);
        acc1[h2 * 4 + t] = __builtin_amdgcn_wmma_f32_16x16x32_bf16(
            false, A1, false, B, (short)0, acc1[h2 * 4 + t], false, false);
      }
    }
  }

  // ---------------- bias + LayerNorm + ReLU (registers) -> LDS ----------------
  float bv[8], gv[8], bev[8];
  #pragma unroll
  for (int t = 0; t < 8; ++t) {
    int n = t * 16 + l16;
    bv[t] = b1[n]; gv[t] = g[n]; bev[t] = be[n];
  }
  ln_relu_to_lds(acc0, lds_a[wave], 0,  bv, gv, bev, l16, hf);
  ln_relu_to_lds(acc1, lds_a[wave], 16, bv, gv, bev, l16, hf);
  __syncthreads();

  // ---------------- GEMM2: h[32 x 128] @ W2[128 x 128] ----------------
  v8f o0[8], o1[8];
  #pragma unroll
  for (int t = 0; t < 8; ++t) { o0[t] = (v8f)0.0f; o1[t] = (v8f)0.0f; }
  #pragma unroll
  for (int c = 0; c < 4; ++c) {
    const int k0 = c * 32 + kb;
    v16bf A0 = __builtin_bit_cast(v16bf,
        seg_pair(*(const uint4*)&lds_a[wave][l16][k0],
                 *(const uint4*)&lds_a[wave][l16][k0 + 16]));
    v16bf A1 = __builtin_bit_cast(v16bf,
        seg_pair(*(const uint4*)&lds_a[wave][16 + l16][k0],
                 *(const uint4*)&lds_a[wave][16 + l16][k0 + 16]));
    const unsigned short* wp = w2s + (size_t)c * (8 * 32 * 16) + (size_t)lane * 16;
    #pragma unroll
    for (int h2 = 0; h2 < 2; ++h2) {
      v8u bu[4];
      #pragma unroll
      for (int t = 0; t < 4; ++t)
        bu[t] = *(const v8u*)(wp + (size_t)(h2 * 4 + t) * (32 * 16));
      #pragma unroll
      for (int t = 0; t < 4; ++t) {
        v16bf B = __builtin_bit_cast(v16bf, bu[t]);
        o0[h2 * 4 + t] = __builtin_amdgcn_wmma_f32_16x16x32_bf16(
            false, A0, false, B, (short)0, o0[h2 * 4 + t], false, false);
        o1[h2 * 4 + t] = __builtin_amdgcn_wmma_f32_16x16x32_bf16(
            false, A1, false, B, (short)0, o1[h2 * 4 + t], false, false);
      }
    }
  }

  // ---------------- bias + store bf16 ----------------
  float b2v[8];
  #pragma unroll
  for (int t = 0; t < 8; ++t) b2v[t] = b2[t * 16 + l16];
  if (rowbase + 32 <= nrows) {                       // fast path: full tile
    #pragma unroll
    for (int t = 0; t < 8; ++t) {
      const int n = t * 16 + l16;
      #pragma unroll
      for (int rr = 0; rr < 8; ++rr) {
        const int row0 = rowbase + hf * 8 + rr;
        out[(size_t)row0 * 128 + n]        = f32_to_bf16(o0[t][rr] + b2v[t]);
        out[(size_t)(row0 + 16) * 128 + n] = f32_to_bf16(o1[t][rr] + b2v[t]);
      }
    }
  } else {
    #pragma unroll
    for (int t = 0; t < 8; ++t) {
      const int n = t * 16 + l16;
      #pragma unroll
      for (int rr = 0; rr < 8; ++rr) {
        int row0 = rowbase + hf * 8 + rr;
        if (row0 < nrows) out[(size_t)row0 * 128 + n] = f32_to_bf16(o0[t][rr] + b2v[t]);
        int row1 = row0 + 16;
        if (row1 < nrows) out[(size_t)row1 * 128 + n] = f32_to_bf16(o1[t][rr] + b2v[t]);
      }
    }
  }
}

// ---------------------------------------------------------------------------
// Helpers: zero, edge scatter-add (msg = segment_sum(e, end)), node scatter,
// count, regressor head.
// ---------------------------------------------------------------------------
__global__ void zero_f32(float* __restrict__ p, int n) {
  int i = blockIdx.x * 256 + threadIdx.x;
  if (i < n) p[i] = 0.f;
}

__global__ void scatter_add_edges(const unsigned short* __restrict__ e,
                                  const int* __restrict__ endI,
                                  float* __restrict__ msg, int E) {
  int id = blockIdx.x * 256 + threadIdx.x;      // E * 32 threads
  int edge = id >> 5, part = id & 31;
  if (edge >= E) return;
  int dst = endI[edge];
  const unsigned short* src = e + (size_t)edge * 128 + part * 4;
  float* d = msg + (size_t)dst * 128 + part * 4;
  uint2 v = *(const uint2*)src;                 // 4 bf16
  atomic_add_f32(d + 0, bf16_to_f32((unsigned short)(v.x & 0xffffu)));
  atomic_add_f32(d + 1, bf16_to_f32((unsigned short)(v.x >> 16)));
  atomic_add_f32(d + 2, bf16_to_f32((unsigned short)(v.y & 0xffffu)));
  atomic_add_f32(d + 3, bf16_to_f32((unsigned short)(v.y >> 16)));
}

__global__ void count_nodes(const int* __restrict__ batch, float* __restrict__ cnt, int N) {
  int i = blockIdx.x * 256 + threadIdx.x;
  if (i < N) atomic_add_f32(&cnt[batch[i]], 1.f);
}

// s[g][0..1023] += xcat[node]  where xcat = concat over 8 layers of x (bf16)
__global__ void scatter_nodes(const unsigned short* __restrict__ xs,
                              const int* __restrict__ batch,
                              float* __restrict__ s, int N) {
  int id = blockIdx.x * 256 + threadIdx.x;      // N * 256 threads
  int node = id >> 8, cg = id & 255;
  if (node >= N) return;
  int gidx = batch[node];
  int col = cg * 4;
  int layer = col >> 7, c = col & 127;
  const unsigned short* src = xs + ((size_t)layer * N + node) * 128 + c;
  uint2 v = *(const uint2*)src;
  float* d = s + (size_t)gidx * 1024 + col;
  atomic_add_f32(d + 0, bf16_to_f32((unsigned short)(v.x & 0xffffu)));
  atomic_add_f32(d + 1, bf16_to_f32((unsigned short)(v.x >> 16)));
  atomic_add_f32(d + 2, bf16_to_f32((unsigned short)(v.y & 0xffffu)));
  atomic_add_f32(d + 3, bf16_to_f32((unsigned short)(v.y >> 16)));
}

__global__ void __launch_bounds__(128)
regressor(const float* __restrict__ s, const float* __restrict__ cnt,
          const float* __restrict__ W1, const float* __restrict__ b1,
          const float* __restrict__ g1, const float* __restrict__ be1,
          const float* __restrict__ W2, const float* __restrict__ b2,
          const float* __restrict__ g2, const float* __restrict__ be2,
          const float* __restrict__ W3, const float* __restrict__ b3,
          float* __restrict__ out, int G, int NB) {
  __shared__ float t_sh[2048];
  __shared__ float h_sh[128];
  const int gidx = blockIdx.x, tid = threadIdx.x;  // 128 threads
  const float inv = 1.f / fmaxf(cnt[gidx], 1.f);
  for (int k = tid; k < 1024; k += 128) {
    float v = s[(size_t)gidx * 1024 + k];
    t_sh[k] = v; t_sh[1024 + k] = v * inv;
  }
  __syncthreads();
  // layer 1: 2048 -> 128
  float acc = b1[tid];
  for (int k = 0; k < 2048; ++k) acc += t_sh[k] * W1[k * 128 + tid];
  h_sh[tid] = acc; __syncthreads();
  float sm = 0.f, sq = 0.f;
  for (int k = 0; k < 128; ++k) { float v = h_sh[k]; sm += v; sq += v * v; }
  __syncthreads();
  {
    float mn = sm * (1.f / 128.f);
    float rs = rsqrtf(sq * (1.f / 128.f) - mn * mn + 1e-5f);
    float v = (acc - mn) * rs * g1[tid] + be1[tid];
    h_sh[tid] = v > 0.f ? v : 0.f;
  }
  __syncthreads();
  // layer 2: 128 -> 128
  float acc2 = b2[tid];
  for (int k = 0; k < 128; ++k) acc2 += h_sh[k] * W2[k * 128 + tid];
  __syncthreads();
  h_sh[tid] = acc2; __syncthreads();
  sm = 0.f; sq = 0.f;
  for (int k = 0; k < 128; ++k) { float v = h_sh[k]; sm += v; sq += v * v; }
  __syncthreads();
  {
    float mn = sm * (1.f / 128.f);
    float rs = rsqrtf(sq * (1.f / 128.f) - mn * mn + 1e-5f);
    float v = (acc2 - mn) * rs * g2[tid] + be2[tid];
    h_sh[tid] = v > 0.f ? v : 0.f;
  }
  __syncthreads();
  // output: 128 -> NB
  if (tid < NB) {
    float o = b3[tid];
    for (int k = 0; k < 128; ++k) o += h_sh[k] * W3[k * NB + tid];
    out[(size_t)gidx * NB + tid] = o;
  }
}

// ---------------------------------------------------------------------------
// Orchestration
// ---------------------------------------------------------------------------
extern "C" void kernel_launch(void* const* d_in, const int* in_sizes, int n_in,
                              void* d_out, int out_size, void* d_ws, size_t ws_size,
                              hipStream_t stream) {
  (void)n_in; (void)ws_size;
  const int SC = 3, D = 128, L = 8, NB = 50;
  const int N = in_sizes[0] / SC;
  const int E = in_sizes[1] / 2;
  const int G = out_size / NB;

  const float* x        = (const float*)d_in[0];
  const int*   eidx     = (const int*)d_in[1];
  const int*   batch    = (const int*)d_in[2];
  const float* enc_W1   = (const float*)d_in[3];
  const float* enc_b1   = (const float*)d_in[4];
  const float* enc_g1   = (const float*)d_in[5];
  const float* enc_be1  = (const float*)d_in[6];
  const float* enc_W2   = (const float*)d_in[7];
  const float* enc_b2   = (const float*)d_in[8];
  const float* eenc_W1  = (const float*)d_in[9];
  const float* eenc_b1  = (const float*)d_in[10];
  const float* eenc_g1  = (const float*)d_in[11];
  const float* eenc_be1 = (const float*)d_in[12];
  const float* eenc_W2  = (const float*)d_in[13];
  const float* eenc_b2  = (const float*)d_in[14];
  const float* nn_W1    = (const float*)d_in[15];
  const float* nn_b1    = (const float*)d_in[16];
  const float* nn_g1    = (const float*)d_in[17];
  const float* nn_be1   = (const float*)d_in[18];
  const float* nn_W2    = (const float*)d_in[19];
  const float* nn_b2    = (const float*)d_in[20];
  const float* en_W1    = (const float*)d_in[21];
  const float* en_b1    = (const float*)d_in[22];
  const float* en_g1    = (const float*)d_in[23];
  const float* en_be1   = (const float*)d_in[24];
  const float* en_W2    = (const float*)d_in[25];
  const float* en_b2    = (const float*)d_in[26];
  const float* reg_W1   = (const float*)d_in[27];
  const float* reg_b1   = (const float*)d_in[28];
  const float* reg_g1   = (const float*)d_in[29];
  const float* reg_be1  = (const float*)d_in[30];
  const float* reg_W2   = (const float*)d_in[31];
  const float* reg_b2   = (const float*)d_in[32];
  const float* reg_g2   = (const float*)d_in[33];
  const float* reg_be2  = (const float*)d_in[34];
  const float* reg_W3   = (const float*)d_in[35];
  const float* reg_b3   = (const float*)d_in[36];

  const int* startI = eidx;
  const int* endI   = eidx + E;

  // ---- carve workspace ----
  char* ws = (char*)d_ws;
  auto carve = [&](size_t bytes) -> void* {
    void* p = (void*)ws;
    ws += (bytes + 255) & ~(size_t)255;
    return p;
  };
  unsigned short* w_eenc1 = (unsigned short*)carve((size_t)256 * 128 * 2);
  unsigned short* w_eenc2 = (unsigned short*)carve((size_t)128 * 128 * 2);
  unsigned short* w_nn1   = (unsigned short*)carve((size_t)L * 256 * 128 * 2);
  unsigned short* w_nn2   = (unsigned short*)carve((size_t)L * 128 * 128 * 2);
  unsigned short* w_en1   = (unsigned short*)carve((size_t)L * 384 * 128 * 2);
  unsigned short* w_en2   = (unsigned short*)carve((size_t)L * 128 * 128 * 2);
  unsigned short* x0      = (unsigned short*)carve((size_t)N * 128 * 2);
  unsigned short* xs      = (unsigned short*)carve((size_t)L * N * 128 * 2);
  unsigned short* ebuf    = (unsigned short*)carve((size_t)E * 128 * 2);
  float*          msg     = (float*)carve((size_t)N * 128 * 4);
  float*          sbuf    = (float*)carve((size_t)G * 1024 * 4);
  float*          cnt     = (float*)carve((size_t)G * 4);

  // ---- shuffle weights into WMMA fragment order (bf16) ----
  auto shuf = [&](const float* W, unsigned short* o, int K) {
    int total = K * 128;
    shuffle_w<<<(total + 255) / 256, 256, 0, stream>>>(W, o, total);
  };
  shuf(eenc_W1, w_eenc1, 256);
  shuf(eenc_W2, w_eenc2, 128);
  for (int i = 0; i < L; ++i) {
    shuf(nn_W1 + (size_t)i * 256 * 128, w_nn1 + (size_t)i * 256 * 128, 256);
    shuf(nn_W2 + (size_t)i * 128 * 128, w_nn2 + (size_t)i * 128 * 128, 128);
    shuf(en_W1 + (size_t)i * 384 * 128, w_en1 + (size_t)i * 384 * 128, 384);
    shuf(en_W2 + (size_t)i * 128 * 128, w_en2 + (size_t)i * 128 * 128, 128);
  }

  // ---- node encoder ----
  node_encoder<<<(N + 7) / 8, 256, 0, stream>>>(x, enc_W1, enc_b1, enc_g1, enc_be1,
                                                enc_W2, enc_b2, x0, N);

  // ---- edge encoder: e = mlp2([x[s], x[d]]) ----
  mlp2_wmma<0, 256><<<(E + 255) / 256, 256, 0, stream>>>(
      E, x0, nullptr, nullptr, startI, endI,
      w_eenc1, eenc_b1, eenc_g1, eenc_be1, w_eenc2, eenc_b2, ebuf);

  // ---- L message-passing layers ----
  const unsigned short* xcur = x0;
  for (int i = 0; i < L; ++i) {
    zero_f32<<<(N * 128 + 255) / 256, 256, 0, stream>>>(msg, N * 128);
    scatter_add_edges<<<(E * 32 + 255) / 256, 256, 0, stream>>>(ebuf, endI, msg, E);
    unsigned short* xout = xs + (size_t)i * N * 128;
    mlp2_wmma<1, 256><<<(N + 255) / 256, 256, 0, stream>>>(
        N, xcur, nullptr, msg, nullptr, nullptr,
        w_nn1 + (size_t)i * 256 * 128, nn_b1 + i * D, nn_g1 + i * D, nn_be1 + i * D,
        w_nn2 + (size_t)i * 128 * 128, nn_b2 + i * D, xout);
    mlp2_wmma<2, 384><<<(E + 255) / 256, 256, 0, stream>>>(
        E, xout, ebuf, nullptr, startI, endI,
        w_en1 + (size_t)i * 384 * 128, en_b1 + i * D, en_g1 + i * D, en_be1 + i * D,
        w_en2 + (size_t)i * 128 * 128, en_b2 + i * D, ebuf);
    xcur = xout;
  }

  // ---- graph aggregation: s = segment_sum(xcat, batch); cnt ----
  zero_f32<<<(G * 1024 + 255) / 256, 256, 0, stream>>>(sbuf, G * 1024);
  zero_f32<<<(G + 255) / 256, 256, 0, stream>>>(cnt, G);
  count_nodes<<<(N + 255) / 256, 256, 0, stream>>>(batch, cnt, N);
  scatter_nodes<<<(N * 256 + 255) / 256, 256, 0, stream>>>(xs, batch, sbuf, N);

  // ---- regressor head ----
  regressor<<<G, 128, 0, stream>>>(sbuf, cnt,
                                   reg_W1, reg_b1, reg_g1, reg_be1,
                                   reg_W2, reg_b2, reg_g2, reg_be2,
                                   reg_W3, reg_b3, (float*)d_out, G, NB);
}
